// TransformerLayer_43937515438608
// MI455X (gfx1250) — matrix operand order repaired and checked by reading, not compiled, not run on hardware
//
#include <hip/hip_runtime.h>
#include <math.h>
#include <stdint.h>

#define B_ 2
#define T_ 2048
#define C_ 1024
#define NH_ 16
#define D_ 64

typedef _Float16 v16h __attribute__((ext_vector_type(16)));
typedef _Float16 v8h  __attribute__((ext_vector_type(8)));
typedef float    v8f  __attribute__((ext_vector_type(8)));
typedef int      vsi4 __attribute__((vector_size(16)));   // matches builtin's V4i pointee

union F16x16 { v16h v; v8h h[2]; };

__device__ __forceinline__ v8f wmma_f16(const F16x16& a, const F16x16& b, v8f c) {
  // D = A(16x32 f16) * B(32x16 f16) + C(16x16 f32)
  return __builtin_amdgcn_wmma_f32_16x16x32_f16(false, a.v, false, b.v, (short)0, c, false, false);
}

__device__ __forceinline__ float selu_f(float x) {
  const float kA = 1.6732632423543772f, kS = 1.0507009873554805f;
  return kS * (x > 0.f ? x : kA * (__expf(x) - 1.f));
}

// ---- CDNA5 async global->LDS copy (16B per lane), ASYNCcnt-tracked ----
__device__ __forceinline__ void async_cp16(const void* g, void* l) {
#if defined(__gfx1250__) && __has_builtin(__builtin_amdgcn_global_load_async_to_lds_b128)
  __builtin_amdgcn_global_load_async_to_lds_b128((vsi4*)g, (vsi4*)l, 0, 0);
#else
  // LDS byte address == low 32 bits of the generic pointer (flat LDS aperture rule)
  unsigned lds = (unsigned)(uintptr_t)l;
  asm volatile("global_load_async_to_lds_b128 %0, %1, off"
               :: "v"(lds), "v"(g) : "memory");
#endif
}

__device__ __forceinline__ void wait_async0() {
#if defined(__gfx1250__) && __has_builtin(__builtin_amdgcn_s_wait_asynccnt)
  __builtin_amdgcn_s_wait_asynccnt(0);
#else
  asm volatile("s_wait_asynccnt 0" ::: "memory");
#endif
}

// ---------------- f32 -> f16 convert ----------------
__global__ void k_f32_to_f16(const float* __restrict__ in, _Float16* __restrict__ out, int n) {
  int i = blockIdx.x * blockDim.x + threadIdx.x;
  if (i < n) out[i] = (_Float16)in[i];
}

// ---------------- AdaLN modulation: cm = selu(c) @ ada_w^T + ada_b ----------------
__global__ __launch_bounds__(256) void k_ada(const float* __restrict__ c,
    const float* __restrict__ ada_w, const float* __restrict__ ada_b,
    float* __restrict__ cm) {
  __shared__ float sc[C_];
  int b = blockIdx.y;
  for (int i = threadIdx.x; i < C_; i += 256) sc[i] = selu_f(c[b * C_ + i]);
  __syncthreads();
  int j = blockIdx.x * 256 + threadIdx.x;   // 0 .. 6C-1
  const float* wr = ada_w + (size_t)j * C_;
  float acc = ada_b[j];
  for (int kk = 0; kk < C_; ++kk) acc += sc[kk] * wr[kk];
  cm[b * 6 * C_ + j] = acc;
}

// ---------------- LayerNorm + (1+scale)*h + shift -> f16 ----------------
__global__ __launch_bounds__(256) void k_ln_mod(const float* __restrict__ xin,
    const float* __restrict__ cm, int shiftOff, int scaleOff,
    _Float16* __restrict__ hout) {
  __shared__ float rs[256], rq[256];
  int row = blockIdx.x;             // b*T + t
  int b = row / T_;
  const float* xr = xin + (size_t)row * C_;
  float x4[4]; float s = 0.f, q = 0.f;
  for (int i = 0; i < 4; ++i) { float v = xr[threadIdx.x + 256 * i]; x4[i] = v; s += v; q += v * v; }
  rs[threadIdx.x] = s; rq[threadIdx.x] = q;
  __syncthreads();
  for (int off = 128; off > 0; off >>= 1) {
    if (threadIdx.x < off) { rs[threadIdx.x] += rs[threadIdx.x + off]; rq[threadIdx.x] += rq[threadIdx.x + off]; }
    __syncthreads();
  }
  float mean = rs[0] * (1.f / C_);
  float var  = rq[0] * (1.f / C_) - mean * mean;
  float rstd = rsqrtf(var + 1e-6f);
  const float* shp = cm + b * 6 * C_ + shiftOff;
  const float* scp = cm + b * 6 * C_ + scaleOff;
  _Float16* ho = hout + (size_t)row * C_;
  for (int i = 0; i < 4; ++i) {
    int ci = threadIdx.x + 256 * i;
    float hn = (x4[i] - mean) * rstd;
    ho[ci] = (_Float16)(hn * (1.f + scp[ci]) + shp[ci]);
  }
}

// ---------------- WMMA GEMM (double-buffered async staging) ----------------
// out = epi(A(MxK,f16) * W(NxK,f16)^T + bias)
// EPI 0: f16 store; EPI 1: selu -> f16; EPI 2: f32 store = resid + gate * v
template<int EPI>
__global__ __launch_bounds__(256) void k_gemm(const _Float16* __restrict__ A,
    const _Float16* __restrict__ W, const float* __restrict__ bias,
    const float* __restrict__ resid, const float* __restrict__ gate,
    void* __restrict__ outp, int M, int N, int K) {
  __shared__ _Float16 sA[2][128][40];
  __shared__ _Float16 sB[2][64][40];
  const int tid  = threadIdx.x;
  const int wave = tid >> 5, lane = tid & 31;
  const int lr = lane & 15, lh = lane >> 4;
  const int wm = wave >> 1, wn = wave & 1;
  const int nBase = blockIdx.x * 64, mBase = blockIdx.y * 128;

  auto stage = [&](int buf, int kt) {
    for (int cchunk = tid; cchunk < 512; cchunk += 256) {
      int r = cchunk >> 2, kc = (cchunk & 3) * 8;
      async_cp16(A + (size_t)(mBase + r) * K + kt + kc, &sA[buf][r][kc]);
    }
    int r = tid >> 2, kc = (tid & 3) * 8;
    async_cp16(W + (size_t)(nBase + r) * K + kt + kc, &sB[buf][r][kc]);
  };

  v8f acc[2][2] = {};
  stage(0, 0);
  wait_async0();
  __syncthreads();
  int cur = 0;
  for (int kt = 0; kt < K; kt += 32) {
    if (kt + 32 < K) stage(cur ^ 1, kt + 32);   // prefetch next tile while computing
    F16x16 af[2], bf[2];
    for (int i = 0; i < 2; ++i) {
      int ar = wm * 32 + i * 16 + lr;
      af[i].h[0] = *(const v8h*)&sA[cur][ar][lh * 8];
      af[i].h[1] = *(const v8h*)&sA[cur][ar][16 + lh * 8];
      int br = wn * 32 + i * 16 + lr;
      bf[i].h[0] = *(const v8h*)&sB[cur][br][lh * 8];
      bf[i].h[1] = *(const v8h*)&sB[cur][br][16 + lh * 8];
    }
    for (int i = 0; i < 2; ++i)
      for (int j = 0; j < 2; ++j)
        acc[i][j] = wmma_f16(af[i], bf[j], acc[i][j]);
    wait_async0();
    __syncthreads();
    cur ^= 1;
  }
  for (int i = 0; i < 2; ++i)
    for (int j = 0; j < 2; ++j) {
      int col = nBase + wn * 32 + j * 16 + lr;
      int m0  = mBase + wm * 32 + i * 16 + lh * 8;
      float bcol = bias[col];
      for (int e = 0; e < 8; ++e) {
        int m = m0 + e;
        float v = acc[i][j][e] + bcol;
        size_t idx = (size_t)m * N + col;
        if (EPI == 0)      ((_Float16*)outp)[idx] = (_Float16)v;
        else if (EPI == 1) ((_Float16*)outp)[idx] = (_Float16)selu_f(v);
        else {
          float g = gate[(m >> 11) * 6 * C_ + col];   // m / T_ (T_=2048)
          ((float*)outp)[idx] = resid[idx] + g * v;
        }
      }
    }
}

// ---------------- RoPE + head split: qkv(b,t,3C) -> q/k/v (b,h,t,d) f16 ----------------
__global__ __launch_bounds__(256) void k_rope(const _Float16* __restrict__ qkv,
    const float* __restrict__ pcos, const float* __restrict__ psin,
    const float* __restrict__ pscl,
    _Float16* __restrict__ q, _Float16* __restrict__ k, _Float16* __restrict__ v) {
  int i = blockIdx.x * 256 + threadIdx.x;   // over B*NH*T*(D/2) = 2^21
  int dp = i & 31;
  int t  = (i >> 5) & (T_ - 1);
  int h  = (i >> 16) & (NH_ - 1);
  int b  = i >> 20;
  int d0 = 2 * dp;
  size_t src = ((size_t)(b * T_ + t)) * (3 * C_) + h * D_ + d0;
  float q0 = (float)qkv[src],          q1 = (float)qkv[src + 1];
  float k0 = (float)qkv[src + C_],     k1 = (float)qkv[src + C_ + 1];
  float v0 = (float)qkv[src + 2 * C_], v1 = (float)qkv[src + 2 * C_ + 1];
  int pi = t * D_ + d0;
  float c0 = pcos[pi], c1 = pcos[pi + 1];
  float s0 = psin[pi], s1 = psin[pi + 1];
  float e0 = pscl[pi], e1 = pscl[pi + 1];
  float qr0 = (q0 * c0 - q1 * s0) * e0;
  float qr1 = (q1 * c1 + q0 * s1) * e1;
  float kr0 = (k0 * c0 - k1 * s0) / e0;
  float kr1 = (k1 * c1 + k0 * s1) / e1;
  size_t dst = ((size_t)((b * NH_ + h) * T_ + t)) * D_ + d0;
  q[dst] = (_Float16)qr0; q[dst + 1] = (_Float16)qr1;
  k[dst] = (_Float16)kr0; k[dst + 1] = (_Float16)kr1;
  v[dst] = (_Float16)v0;  v[dst + 1] = (_Float16)v1;
}

// ---------------- Flash attention (causal), WMMA QK^T and PV ----------------
__global__ __launch_bounds__(256) void k_attn(const _Float16* __restrict__ q,
    const _Float16* __restrict__ k, const _Float16* __restrict__ v,
    float* __restrict__ attnf) {
  __shared__ _Float16 sK[32][72];       // [kpos][d]
  __shared__ _Float16 sVt[64][40];      // [d][kpos]
  __shared__ _Float16 sP[8][16][40];    // per-wave P tile [row][kpos]
  const int tid  = threadIdx.x;
  const int wave = tid >> 5, lane = tid & 31;
  const int lr = lane & 15, lh = lane >> 4;
  const int nq = T_ / 128;
  const int qb = blockIdx.x % nq;
  const int bh = blockIdx.x / nq;       // b*NH + h
  const int h = bh % NH_, b = bh / NH_;
  const int qbase = qb * 128;
  const int qrow0 = qbase + wave * 16;
  const size_t headOff = (size_t)bh * T_ * D_;
  F16x16 qa[2];
  {
    const _Float16* qp = q + headOff + (size_t)(qrow0 + lr) * D_;
    for (int dk = 0; dk < 2; ++dk) {
      qa[dk].h[0] = *(const v8h*)(qp + dk * 32 + lh * 8);
      qa[dk].h[1] = *(const v8h*)(qp + dk * 32 + 16 + lh * 8);
    }
  }
  v8f o[4] = {};
  v8f mrun, lrun;
  for (int e = 0; e < 8; ++e) { mrun[e] = -1e30f; lrun[e] = 0.f; }
  const float inv_tp = 0.08838834764831845f;    // 1/sqrt(2*D)
  const int kbMax = (qbase + 127) >> 5;
  for (int kb = 0; kb <= kbMax; ++kb) {
    { // stage K tile via async DMA to LDS; V^T via register transpose
      int r = tid >> 3, cc = (tid & 7) * 8;
      async_cp16(k + headOff + (size_t)(kb * 32 + r) * D_ + cc, &sK[r][cc]);
      const v8h vv = *(const v8h*)(v + headOff + (size_t)(kb * 32 + r) * D_ + cc);
      for (int e = 0; e < 8; ++e) sVt[cc + e][r] = vv[e];
    }
    wait_async0();
    __syncthreads();
    v8f st[2];
    for (int nt = 0; nt < 2; ++nt) {
      v8f s = {};
      for (int dk = 0; dk < 2; ++dk) {
        F16x16 bb;
        bb.h[0] = *(const v8h*)&sK[nt * 16 + lr][dk * 32 + lh * 8];
        bb.h[1] = *(const v8h*)&sK[nt * 16 + lr][dk * 32 + 16 + lh * 8];
        s = wmma_f16(qa[dk], bb, s);
      }
      int kpos = kb * 32 + nt * 16 + lr;
      for (int e = 0; e < 8; ++e) {
        int rowg = qrow0 + lh * 8 + e;
        st[nt][e] = (kpos <= rowg) ? s[e] * inv_tp : -1e30f;
      }
    }
    v8f bm;
    for (int e = 0; e < 8; ++e) bm[e] = fmaxf(st[0][e], st[1][e]);
    for (int off = 1; off < 16; off <<= 1)
      for (int e = 0; e < 8; ++e) bm[e] = fmaxf(bm[e], __shfl_xor(bm[e], off, 32));
    v8f mnew, alpha;
    for (int e = 0; e < 8; ++e) {
      mnew[e]  = fmaxf(mrun[e], bm[e]);
      alpha[e] = __expf(mrun[e] - mnew[e]);
      mrun[e]  = mnew[e];
    }
    v8f rssum;
    for (int e = 0; e < 8; ++e) {
      st[0][e] = __expf(st[0][e] - mnew[e]);
      st[1][e] = __expf(st[1][e] - mnew[e]);
      rssum[e] = st[0][e] + st[1][e];
    }
    for (int off = 1; off < 16; off <<= 1)
      for (int e = 0; e < 8; ++e) rssum[e] += __shfl_xor(rssum[e], off, 32);
    for (int e = 0; e < 8; ++e) {
      lrun[e] = lrun[e] * alpha[e] + rssum[e];
      for (int dt = 0; dt < 4; ++dt) o[dt][e] *= alpha[e];
    }
    for (int nt = 0; nt < 2; ++nt)
      for (int e = 0; e < 8; ++e)
        sP[wave][lh * 8 + e][nt * 16 + lr] = (_Float16)st[nt][e];
    F16x16 pa;
    pa.h[0] = *(const v8h*)&sP[wave][lr][lh * 8];
    pa.h[1] = *(const v8h*)&sP[wave][lr][16 + lh * 8];
    for (int dt = 0; dt < 4; ++dt) {
      F16x16 bb;
      bb.h[0] = *(const v8h*)&sVt[dt * 16 + lr][lh * 8];
      bb.h[1] = *(const v8h*)&sVt[dt * 16 + lr][16 + lh * 8];
      o[dt] = wmma_f16(pa, bb, o[dt]);
    }
    __syncthreads();
  }
  for (int dt = 0; dt < 4; ++dt)
    for (int e = 0; e < 8; ++e) {
      int rowg = qrow0 + lh * 8 + e;
      attnf[((size_t)b * T_ + rowg) * C_ + h * D_ + dt * 16 + lr] = o[dt][e] / lrun[e];
    }
}

// ---------------- x1 = x + gate * attn ----------------
__global__ void k_resid(const float* __restrict__ x, const float* __restrict__ attnf,
    const float* __restrict__ cm, float* __restrict__ out) {
  int i = blockIdx.x * 256 + threadIdx.x;   // B*T*C = 2^22
  int ci = i & (C_ - 1);
  int b  = i >> 21;                          // / (T_*C_)
  out[i] = x[i] + cm[b * 6 * C_ + 2 * C_ + ci] * attnf[i];
}

extern "C" void kernel_launch(void* const* d_in, const int* in_sizes, int n_in,
                              void* d_out, int out_size, void* d_ws, size_t ws_size,
                              hipStream_t stream) {
  (void)in_sizes; (void)n_in; (void)out_size; (void)ws_size;
  const float* x      = (const float*)d_in[0];
  const float* pe_cos = (const float*)d_in[1];
  const float* pe_sin = (const float*)d_in[2];
  const float* pe_scl = (const float*)d_in[3];
  const float* c      = (const float*)d_in[4];
  /* d_in[5] = mask (causal, computed analytically) */
  const float* qkv_w  = (const float*)d_in[6];
  const float* qkv_b  = (const float*)d_in[7];
  const float* w1     = (const float*)d_in[8];
  const float* b1     = (const float*)d_in[9];
  const float* w2     = (const float*)d_in[10];
  const float* b2     = (const float*)d_in[11];
  const float* ada_w  = (const float*)d_in[12];
  const float* ada_b  = (const float*)d_in[13];
  float* out = (float*)d_out;

  char* ws = (char*)d_ws;
  size_t off = 0;
  auto alloc = [&](size_t bytes) { char* p = ws + off; off += (bytes + 255) & ~(size_t)255; return p; };
  _Float16* wq16  = (_Float16*)alloc((size_t)3 * C_ * C_ * 2);
  _Float16* w116  = (_Float16*)alloc((size_t)4 * C_ * C_ * 2);
  _Float16* w216  = (_Float16*)alloc((size_t)4 * C_ * C_ * 2);
  float*    cm    = (float*)   alloc((size_t)B_ * 6 * C_ * 4);
  _Float16* h16   = (_Float16*)alloc((size_t)B_ * T_ * C_ * 2);
  _Float16* qkv16 = (_Float16*)alloc((size_t)B_ * T_ * 3 * C_ * 2);
  _Float16* q16   = (_Float16*)alloc((size_t)B_ * NH_ * T_ * D_ * 2);
  _Float16* k16   = (_Float16*)alloc((size_t)B_ * NH_ * T_ * D_ * 2);
  _Float16* v16   = (_Float16*)alloc((size_t)B_ * NH_ * T_ * D_ * 2);
  float*    attnf = (float*)   alloc((size_t)B_ * T_ * C_ * 4);
  _Float16* ff16  = (_Float16*)alloc((size_t)B_ * T_ * 4 * C_ * 2);

  int n;
  n = 3 * C_ * C_; k_f32_to_f16<<<(n + 255) / 256, 256, 0, stream>>>(qkv_w, wq16, n);
  n = 4 * C_ * C_; k_f32_to_f16<<<(n + 255) / 256, 256, 0, stream>>>(w1,    w116, n);
  n = 4 * C_ * C_; k_f32_to_f16<<<(n + 255) / 256, 256, 0, stream>>>(w2,    w216, n);

  k_ada<<<dim3(6 * C_ / 256, B_), 256, 0, stream>>>(c, ada_w, ada_b, cm);

  // attention branch
  k_ln_mod<<<B_ * T_, 256, 0, stream>>>(x, cm, 0, C_, h16);
  k_gemm<0><<<dim3(3 * C_ / 64, B_ * T_ / 128), 256, 0, stream>>>(
      h16, wq16, qkv_b, nullptr, nullptr, qkv16, B_ * T_, 3 * C_, C_);
  k_rope<<<(B_ * NH_ * T_ * (D_ / 2)) / 256, 256, 0, stream>>>(
      qkv16, pe_cos, pe_sin, pe_scl, q16, k16, v16);
  k_attn<<<B_ * NH_ * (T_ / 128), 256, 0, stream>>>(q16, k16, v16, attnf);
  k_resid<<<(B_ * T_ * C_) / 256, 256, 0, stream>>>(x, attnf, cm, out);

  // FFN branch
  k_ln_mod<<<B_ * T_, 256, 0, stream>>>(out, cm, 3 * C_, 4 * C_, h16);
  k_gemm<1><<<dim3(4 * C_ / 64, B_ * T_ / 128), 256, 0, stream>>>(
      h16, w116, b1, nullptr, nullptr, ff16, B_ * T_, 4 * C_, C_);
  k_gemm<2><<<dim3(C_ / 64, B_ * T_ / 128), 256, 0, stream>>>(
      ff16, w216, b2, out, cm + 5 * C_, out, B_ * T_, C_, 4 * C_);
}
